// get_semantic_model_36404142801662
// MI455X (gfx1250) — matrix operand order repaired and checked by reading, not compiled, not run on hardware
//
#include <hip/hip_runtime.h>
#include <hip/hip_bf16.h>

typedef _Float16 h16 __attribute__((ext_vector_type(16)));
typedef _Float16 h8  __attribute__((ext_vector_type(8)));
typedef float    f8  __attribute__((ext_vector_type(8)));

#define B_    4
#define N_    16384
#define S_    4096
#define D1_   64
#define D2_   128
#define CIN_  192
#define COUT_ 128

// ---------------------------------------------------------------------------
// Weight conversion f32 -> f16 (weights are tiny; L2-resident thereafter)
// ---------------------------------------------------------------------------
__global__ __launch_bounds__(256) void cvt_weights_kernel(
    const float* __restrict__ wf, const float* __restrict__ w1,
    const float* __restrict__ w2, const float* __restrict__ w3,
    const float* __restrict__ w4, _Float16* __restrict__ dst)
{
    int i = blockIdx.x * 256 + threadIdx.x;
    const int F = COUT_ * CIN_;    // 24576
    const int Q = COUT_ * COUT_;   // 16384
    if (i < F)            dst[i] = (_Float16)wf[i];
    else if (i < F + Q)   dst[i] = (_Float16)w1[i - F];
    else if (i < F + 2*Q) dst[i] = (_Float16)w2[i - F - Q];
    else if (i < F + 3*Q) dst[i] = (_Float16)w3[i - F - 2*Q];
    else if (i < F + 4*Q) dst[i] = (_Float16)w4[i - F - 3*Q];
}

// ---------------------------------------------------------------------------
// points2 [B][D2][S] f32 -> P2T [B][S][D2] f16  (so KNN gather is contiguous)
// ---------------------------------------------------------------------------
__global__ __launch_bounds__(256) void cvt_points2_kernel(
    const float* __restrict__ p2, _Float16* __restrict__ P2T)
{
    long i = (long)blockIdx.x * 256 + threadIdx.x;   // < B*D2*S, coalesced read
    int s = (int)(i % S_);
    int d = (int)((i / S_) % D2_);
    int b = (int)(i / ((long)S_ * D2_));
    P2T[((size_t)b * S_ + s) * D2_ + d] = (_Float16)p2[i];
}

// ---------------------------------------------------------------------------
// points1 [B][64][N] f32 -> cols 0..63 of Xin [B][N][192] f16
// One thread per point: reads coalesced across threads per channel.
// ---------------------------------------------------------------------------
__global__ __launch_bounds__(256) void cvt_points1_kernel(
    const float* __restrict__ p1, _Float16* __restrict__ Xin)
{
    int t = blockIdx.x * 256 + threadIdx.x;          // < B*N
    int n = t % N_;
    int b = t / N_;
    const float* src = p1 + (size_t)b * D1_ * N_ + n;
    _Float16* xo = Xin + ((size_t)b * N_ + n) * CIN_;
    #pragma unroll 8
    for (int c = 0; c < D1_; ++c)
        xo[c] = (_Float16)src[(size_t)c * N_];
}

// ---------------------------------------------------------------------------
// Brute-force 3-NN + inverse-distance interpolation.
// xyz2[b] (4096x3 = 48KB) staged fully in LDS; inner loop is LDS+VALU only.
// Gathers from P2T [B][S][D2] (3 contiguous 256B rows), writes cols 64..191
// of Xin [B][N][192] as 16-B vector stores.
// ---------------------------------------------------------------------------
__global__ __launch_bounds__(256) void knn_interp_kernel(
    const float* __restrict__ xyz1, const float* __restrict__ xyz2,
    const _Float16* __restrict__ P2T, _Float16* __restrict__ Xin)
{
    __shared__ float sx[S_], sy[S_], sz[S_];          // 48 KB
    int b   = blockIdx.y;
    int tid = threadIdx.x;
    for (int i = tid; i < S_; i += 256) {
        const float* p = xyz2 + ((size_t)b * S_ + i) * 3;
        sx[i] = p[0]; sy[i] = p[1]; sz[i] = p[2];
    }
    __syncthreads();

    int n = blockIdx.x * 256 + tid;
    const float* q = xyz1 + ((size_t)b * N_ + n) * 3;
    float px = q[0], py = q[1], pz = q[2];

    float d0 = 3.4e38f, d1 = 3.4e38f, d2 = 3.4e38f;
    int   i0 = 0, i1 = 0, i2 = 0;
    for (int s = 0; s < S_; ++s) {
        float dx = sx[s] - px, dy = sy[s] - py, dz = sz[s] - pz;
        float d  = dx*dx + dy*dy + dz*dz;
        if (d < d0)      { d2 = d1; i2 = i1; d1 = d0; i1 = i0; d0 = d; i0 = s; }
        else if (d < d1) { d2 = d1; i2 = i1; d1 = d;  i1 = s; }
        else if (d < d2) { d2 = d;  i2 = s; }
    }
    float r0 = 1.f/(d0 + 1e-8f), r1 = 1.f/(d1 + 1e-8f), r2 = 1.f/(d2 + 1e-8f);
    float rs = r0 + r1 + r2;
    float w0 = r0/rs, w1 = r1/rs, w2 = r2/rs;

    const _Float16* g0 = P2T + ((size_t)b * S_ + i0) * D2_;
    const _Float16* g1 = P2T + ((size_t)b * S_ + i1) * D2_;
    const _Float16* g2 = P2T + ((size_t)b * S_ + i2) * D2_;
    _Float16* xo = Xin + ((size_t)b * N_ + n) * CIN_ + D1_;
    #pragma unroll
    for (int c8 = 0; c8 < D2_ / 8; ++c8) {
        h8 a0 = *reinterpret_cast<const h8*>(g0 + c8 * 8);
        h8 a1 = *reinterpret_cast<const h8*>(g1 + c8 * 8);
        h8 a2 = *reinterpret_cast<const h8*>(g2 + c8 * 8);
        h8 o;
        #pragma unroll
        for (int j = 0; j < 8; ++j)
            o[j] = (_Float16)(w0 * (float)a0[j] + w1 * (float)a1[j]
                              + w2 * (float)a2[j]);
        *reinterpret_cast<h8*>(xo + c8 * 8) = o;
    }
}

// ---------------------------------------------------------------------------
// Y = lrelu(W @ X + b [+ Res]) via v_wmma_f32_16x16x32_f16.
// Activations are point-major [B][N][C]:
//   - X tile staging = straight row copy (global b128 -> ds b128, no transpose)
//   - epilogue: one 16-B f16 store / 16-B residual load per lane
// K templated so all fragment loops unroll and aFrag stays in VGPRs.
// Block: 256 threads = 8 waves; wave w owns cout rows [16w,16w+16) x 64 pts.
// ---------------------------------------------------------------------------
template <int K, bool HAS_RES, bool OUT16, bool OUT32>
__global__ __launch_bounds__(256) void gemm_lrelu_wmma_kernel(
    const _Float16* __restrict__ Wf16,   // [COUT][K]
    const float*    __restrict__ bias,   // [COUT]
    const _Float16* __restrict__ X,      // [B][N][K]
    const _Float16* __restrict__ Res,    // [B][N][COUT]
    _Float16*       __restrict__ out16,  // [B][N][COUT]
    float*          __restrict__ out32)  // [B][COUT][N]  (reference layout)
{
    constexpr int NK = K / 32;           // 6 (K=192) or 4 (K=128)
    __shared__ _Float16 sXT[64][200];    // X tile rows, padded (25.6 KB)

    int b   = blockIdx.y;
    int n0  = blockIdx.x * 64;
    int tid = threadIdx.x;

    // Stage 64 point-rows of K halfs: pure 16-B copies, coalesced.
    const _Float16* Xb = X + ((size_t)b * N_ + n0) * K;
    #pragma unroll
    for (int it = 0; it < (64 * K / 8) / 256; ++it) {
        int chunk = it * 256 + tid;
        int nl = chunk / (K / 8);
        int kk = (chunk % (K / 8)) * 8;
        *reinterpret_cast<h8*>(&sXT[nl][kk]) =
            *reinterpret_cast<const h8*>(Xb + (size_t)nl * K + kk);
    }
    __syncthreads();

    int wid  = tid >> 5, lane = tid & 31;
    int row  = lane & 15, hi = lane >> 4;           // hi: lane-group 0/1
    int co_base = wid * 16;

    // A-fragments (16x32 f16): lane L holds row M=L%16; halfs 0..7 -> K=Loff+0..7,
    // halfs 8..15 -> K=16+Loff+0..7, Loff = 8*(L>=16). Two 16B loads per chunk.
    h16 aFrag[NK];
    const _Float16* wrow = Wf16 + (size_t)(co_base + row) * K;
    #pragma unroll
    for (int kc = 0; kc < NK; ++kc) {
        h8 lo  = *reinterpret_cast<const h8*>(wrow + kc * 32 + hi * 8);
        h8 hiP = *reinterpret_cast<const h8*>(wrow + kc * 32 + 16 + hi * 8);
        aFrag[kc] = __builtin_shufflevector(lo, hiP,
            0,1,2,3,4,5,6,7,8,9,10,11,12,13,14,15);
    }

    // Bias for this lane's 8 output channels (co = co_base + hi*8 + v).
    float bv[8];
    #pragma unroll
    for (int v = 0; v < 8; ++v) bv[v] = bias[co_base + hi * 8 + v];

    #pragma unroll
    for (int nt = 0; nt < 4; ++nt) {
        int nl = nt * 16 + row;                     // point col = lane%16 in tile
        f8 c = {};
        #pragma unroll
        for (int kc = 0; kc < NK; ++kc) {
            // B-fragment (32x16 f16): lane L holds col N=L%16; halfs 0..7 -> K=8*hi+j, halfs 8..15 -> K=16+8*hi+j
            const _Float16* bp = &sXT[nl][kc * 32 + hi * 8];
            h8 b0 = *reinterpret_cast<const h8*>(bp);
            h8 b1 = *reinterpret_cast<const h8*>(bp + 16);
            h16 bf = __builtin_shufflevector(b0, b1,
                0,1,2,3,4,5,6,7,8,9,10,11,12,13,14,15);
            c = __builtin_amdgcn_wmma_f32_16x16x32_f16(
                false, aFrag[kc], false, bf, (short)0, c, false, false);
        }
        int n = n0 + nl;
        size_t rowo = ((size_t)b * N_ + n) * COUT_ + co_base + hi * 8;
        float y[8];
        #pragma unroll
        for (int v = 0; v < 8; ++v) y[v] = c[v] + bv[v];
        if constexpr (HAS_RES) {
            h8 r = *reinterpret_cast<const h8*>(Res + rowo);
            #pragma unroll
            for (int v = 0; v < 8; ++v) y[v] += (float)r[v];
        }
        #pragma unroll
        for (int v = 0; v < 8; ++v) y[v] = (y[v] > 0.f) ? y[v] : 0.01f * y[v];
        if constexpr (OUT16) {
            h8 o;
            #pragma unroll
            for (int v = 0; v < 8; ++v) o[v] = (_Float16)y[v];
            *reinterpret_cast<h8*>(out16 + rowo) = o;   // one b128 store
        }
        if constexpr (OUT32) {
            #pragma unroll
            for (int v = 0; v < 8; ++v) {
                int co = co_base + hi * 8 + v;
                out32[((size_t)b * COUT_ + co) * N_ + n] = y[v];
            }
        }
    }
}

// ---------------------------------------------------------------------------
extern "C" void kernel_launch(void* const* d_in, const int* in_sizes, int n_in,
                              void* d_out, int out_size, void* d_ws, size_t ws_size,
                              hipStream_t stream)
{
    (void)in_sizes; (void)n_in; (void)out_size; (void)ws_size;
    const float* xyz1    = (const float*)d_in[0];
    const float* xyz2    = (const float*)d_in[1];
    const float* points1 = (const float*)d_in[2];
    const float* points2 = (const float*)d_in[3];
    const float* W_fuse  = (const float*)d_in[4];
    const float* b_fuse  = (const float*)d_in[5];
    const float* W1 = (const float*)d_in[6];
    const float* b1 = (const float*)d_in[7];
    const float* W2 = (const float*)d_in[8];
    const float* b2 = (const float*)d_in[9];
    const float* W3 = (const float*)d_in[10];
    const float* b3 = (const float*)d_in[11];
    const float* W4 = (const float*)d_in[12];
    const float* b4 = (const float*)d_in[13];
    float* out = (float*)d_out;

    // workspace layout (all f16): weights | P2T | Xin | X0 | X1 | H
    char* ws = (char*)d_ws;
    _Float16* wAll = (_Float16*)ws;                        // 90112 halfs
    size_t off = 256 * 1024;
    _Float16* P2T  = (_Float16*)(ws + off); off += (size_t)B_ * S_ * D2_ * 2;
    _Float16* Xin  = (_Float16*)(ws + off); off += (size_t)B_ * N_ * CIN_  * 2;
    _Float16* X0   = (_Float16*)(ws + off); off += (size_t)B_ * N_ * COUT_ * 2;
    _Float16* X1   = (_Float16*)(ws + off); off += (size_t)B_ * N_ * COUT_ * 2;
    _Float16* Hbuf = (_Float16*)(ws + off);

    _Float16* Wfuse16 = wAll;
    _Float16* W1_16 = wAll + COUT_ * CIN_;
    _Float16* W2_16 = W1_16 + COUT_ * COUT_;
    _Float16* W3_16 = W2_16 + COUT_ * COUT_;
    _Float16* W4_16 = W3_16 + COUT_ * COUT_;

    int nW = COUT_ * CIN_ + 4 * COUT_ * COUT_;             // 90112
    cvt_weights_kernel<<<(nW + 255) / 256, 256, 0, stream>>>(
        W_fuse, W1, W2, W3, W4, wAll);
    cvt_points2_kernel<<<((long)B_ * D2_ * S_) / 256, 256, 0, stream>>>(
        points2, P2T);
    cvt_points1_kernel<<<(B_ * N_) / 256, 256, 0, stream>>>(points1, Xin);
    knn_interp_kernel<<<dim3(N_ / 256, B_), 256, 0, stream>>>(
        xyz1, xyz2, P2T, Xin);

    dim3 g(N_ / 64, B_);
    // x0 = lrelu(Wfuse @ Xin + bfuse)
    gemm_lrelu_wmma_kernel<CIN_,  false, true,  false><<<g, 256, 0, stream>>>(
        Wfuse16, b_fuse, Xin, nullptr, X0, nullptr);
    // h1 = lrelu(W1 @ x0 + b1)
    gemm_lrelu_wmma_kernel<COUT_, false, true,  false><<<g, 256, 0, stream>>>(
        W1_16, b1, X0, nullptr, Hbuf, nullptr);
    // x1 = lrelu(W2 @ h1 + b2 + x0)
    gemm_lrelu_wmma_kernel<COUT_, true,  true,  false><<<g, 256, 0, stream>>>(
        W2_16, b2, Hbuf, X0, X1, nullptr);
    // h2 = lrelu(W3 @ x1 + b3)
    gemm_lrelu_wmma_kernel<COUT_, false, true,  false><<<g, 256, 0, stream>>>(
        W3_16, b3, X1, nullptr, Hbuf, nullptr);
    // out = lrelu(W4 @ h2 + b4 + x1)   (f32 output, reference [B][CO][N] layout)
    gemm_lrelu_wmma_kernel<COUT_, true,  false, true><<<g, 256, 0, stream>>>(
        W4_16, b4, Hbuf, X1, nullptr, out);
}
